// MHA_34050500723480
// MI455X (gfx1250) — compile-verified, matrix operand order checked by reading, not run on hardware
//
#include <hip/hip_runtime.h>
#include <stdint.h>

// ---------------------------------------------------------------------------
// CDNA5 (gfx1250) wave32 WMMA MHA forward.
//   B=2, S=2048, E=2048, H=16, D=128.  M = B*S = 4096.
// v_wmma_f32_16x16x32_bf16 everywhere; LDS staged with double-buffered
// global_load_async_to_lds_b128 (ASYNCcnt); fragments = 2x ds_load_b128.
// ---------------------------------------------------------------------------

typedef __attribute__((ext_vector_type(16))) __bf16   v16bf;
typedef __attribute__((ext_vector_type(8)))  float    v8f;
typedef __attribute__((ext_vector_type(4)))  unsigned u32x4;
typedef __attribute__((ext_vector_type(2)))  unsigned u32x2;
typedef __attribute__((ext_vector_type(4)))  float    f32x4;

struct U4x2 { u32x4 lo, hi; };

#define WMMA_BF16(A, B, C) \
  __builtin_amdgcn_wmma_f32_16x16x32_bf16(false, (A), false, (B), (short)0, (C), false, false)

__device__ __forceinline__ unsigned bfbits(float f) {
  unsigned u = __builtin_bit_cast(unsigned, f);
  return ((u + 0x7FFFu + ((u >> 16) & 1u)) >> 16) & 0xFFFFu;  // RNE
}
__device__ __forceinline__ __bf16 f2bf(float f) {
  unsigned short h = (unsigned short)bfbits(f);
  return __builtin_bit_cast(__bf16, h);
}

// 16x32 16-bit fragment, K-contiguous source rows (src[m*ld + k]):
// lane L<16 -> row M=L, K={0..7,16..23}; lane>=16 -> row M=L-16, K={8..15,24..31}.
__device__ __forceinline__ v16bf load_frag_row16(const __bf16* src, int ld, int lane) {
  int m  = lane & 15;
  int kb = (lane >> 4) << 3;
  const __bf16* p = src + (size_t)m * ld + kb;
  U4x2 t;
  t.lo = *reinterpret_cast<const u32x4*>(p);
  t.hi = *reinterpret_cast<const u32x4*>(p + 16);
  return __builtin_bit_cast(v16bf, t);
}

// Per-lane async 16B global -> LDS copy (GLOBAL_LOAD_ASYNC_TO_LDS_B128, ASYNCcnt).
__device__ __forceinline__ void async_copy16(unsigned lds_off, const void* gsrc) {
  unsigned long long g = (unsigned long long)(uintptr_t)gsrc;
  asm volatile("global_load_async_to_lds_b128 %0, %1, off"
               :: "v"(lds_off), "v"(g) : "memory");
}
__device__ __forceinline__ void wait_async0() {
  asm volatile("s_wait_asynccnt 0x0" ::: "memory");
}
__device__ __forceinline__ unsigned lds_addr(const void* p) {
  return (unsigned)(uintptr_t)p;
}

// ---------------------------------------------------------------------------
// fp32 -> bf16, 4 elements / thread (row-major copy for x).
// ---------------------------------------------------------------------------
__global__ __launch_bounds__(256) void cvt4(const f32x4* __restrict__ in,
                                            u32x2* __restrict__ out, int n4) {
  int i = blockIdx.x * blockDim.x + threadIdx.x;
  if (i >= n4) return;
  f32x4 f = in[i];
  u32x2 p;
  p.x = bfbits(f.x) | (bfbits(f.y) << 16);
  p.y = bfbits(f.z) | (bfbits(f.w) << 16);
  out[i] = p;
}

// ---------------------------------------------------------------------------
// Fused fp32->bf16 + transpose: in [K][N] f32 -> out [N][K] bf16. 32x32 tiles.
// ---------------------------------------------------------------------------
__global__ __launch_bounds__(256) void cvtT(const float* __restrict__ in,
                                            __bf16* __restrict__ out, int K, int N) {
  __shared__ __align__(16) __bf16 tile[32][40];
  int tid = threadIdx.x;
  int k0 = blockIdx.y * 32, n0 = blockIdx.x * 32;
  int row = tid >> 3, seg = tid & 7;

  f32x4 f = *reinterpret_cast<const f32x4*>(in + (size_t)(k0 + row) * N + n0 + seg * 4);
  u32x2 p;
  p.x = bfbits(f.x) | (bfbits(f.y) << 16);
  p.y = bfbits(f.z) | (bfbits(f.w) << 16);
  *reinterpret_cast<u32x2*>(&tile[row][seg * 4]) = p;
  __syncthreads();

  unsigned a = __builtin_bit_cast(unsigned short, tile[seg * 4 + 0][row]);
  unsigned b = __builtin_bit_cast(unsigned short, tile[seg * 4 + 1][row]);
  unsigned c = __builtin_bit_cast(unsigned short, tile[seg * 4 + 2][row]);
  unsigned d = __builtin_bit_cast(unsigned short, tile[seg * 4 + 3][row]);
  u32x2 q;
  q.x = a | (b << 16);
  q.y = c | (d << 16);
  *reinterpret_cast<u32x2*>(out + (size_t)(n0 + row) * K + k0 + seg * 4) = q;
}

// ---------------------------------------------------------------------------
// Tiled bf16 GEMM: C[M,N] = A[M,K] * Bt[N,K]^T + bias.
// Block tile 128x128, 8 waves (4M x 2N), wave tile 32x64.
// Double-buffered async LDS staging; one barrier per k-step.
// mode 0: fp32 store to outF.
// mode 1: QKV split -> Q,K bf16 [B,H,S,D]; V bf16 [B,H,D,S].
// ---------------------------------------------------------------------------
__global__ __launch_bounds__(256) void gemm_bf16_wmma(
    const __bf16* __restrict__ A, const __bf16* __restrict__ Bt,
    const float* __restrict__ bias, float* __restrict__ outF,
    __bf16* __restrict__ qb, __bf16* __restrict__ kbuf, __bf16* __restrict__ vb,
    int M, int N, int K, int mode) {
  __shared__ __align__(16) __bf16 As[2][128][32];
  __shared__ __align__(16) __bf16 Bts[2][128][32];

  int tid  = threadIdx.x;
  int lane = tid & 31;
  int wid  = tid >> 5;
  int wm   = (wid >> 1) * 32;  // 0,32,64,96
  int wn   = (wid & 1) * 64;   // 0 / 64
  int m0   = blockIdx.y * 128;
  int n0   = blockIdx.x * 128;

  unsigned ldsA[2] = { lds_addr(&As[0][0][0]), lds_addr(&As[1][0][0]) };
  unsigned ldsB[2] = { lds_addr(&Bts[0][0][0]), lds_addr(&Bts[1][0][0]) };

  // Stage one 128x32 A tile + 128x32 Bt tile: 1024 x 16B, 4 copies / thread.
  auto stage = [&](int buf, int k0) {
#pragma unroll
    for (int t = 0; t < 2; ++t) {
      int idx = tid + t * 256;
      int row = idx >> 2, seg = idx & 3;
      async_copy16(ldsA[buf] + idx * 16, A + (size_t)(m0 + row) * K + k0 + seg * 8);
      async_copy16(ldsB[buf] + idx * 16, Bt + (size_t)(n0 + row) * K + k0 + seg * 8);
    }
  };
  stage(0, 0);

  v8f acc[2][4] = {};
  int nsteps = K >> 5;

  for (int s = 0; s < nsteps; ++s) {
    int cur = s & 1;
    wait_async0();       // our copies for buf[cur] have landed
    __syncthreads();     // everyone's have
    if (s + 1 < nsteps) stage(cur ^ 1, (s + 1) * 32);
    if (s + 2 < nsteps) {  // warm L2 two steps ahead (global_prefetch_b8)
      __builtin_prefetch(A + (size_t)(m0 + (tid >> 2)) * K + (s + 2) * 32, 0, 1);
      __builtin_prefetch(Bt + (size_t)(n0 + (tid >> 2)) * K + (s + 2) * 32, 0, 1);
    }

    v16bf af0 = load_frag_row16(&As[cur][wm][0], 32, lane);
    v16bf af1 = load_frag_row16(&As[cur][wm + 16][0], 32, lane);
#pragma unroll
    for (int nt = 0; nt < 4; ++nt) {
      v16bf bfrag = load_frag_row16(&Bts[cur][wn + nt * 16][0], 32, lane);
      acc[0][nt] = WMMA_BF16(af0, bfrag, acc[0][nt]);
      acc[1][nt] = WMMA_BF16(af1, bfrag, acc[1][nt]);
    }
  }

  // C/D layout: vgpr r, lane l -> M = r + 8*(l>>4), N = l&15.
  int lh = lane >> 4, nl = lane & 15;
#pragma unroll
  for (int mi = 0; mi < 2; ++mi) {
#pragma unroll
    for (int nt = 0; nt < 4; ++nt) {
#pragma unroll
      for (int r = 0; r < 8; ++r) {
        int row = m0 + wm + mi * 16 + r + 8 * lh;
        int col = n0 + wn + nt * 16 + nl;
        float v = acc[mi][nt][r] + bias[col];
        if (mode == 0) {
          outF[(size_t)row * N + col] = v;
        } else {
          int which = col >> 11;  // 0=q 1=k 2=v
          int e = col & 2047;
          int h = e >> 7, d = e & 127;
          int b = row >> 11, ss = row & 2047;
          if (which == 2) {  // V: [B,H,D,S]
            vb[((((size_t)b * 16 + h) * 128) + d) * 2048 + ss] = f2bf(v);
          } else {           // Q,K: [B,H,S,D]
            __bf16* dst = (which == 0) ? qb : kbuf;
            dst[((((size_t)b * 16 + h) * 2048) + ss) * 128 + d] = f2bf(v);
          }
        }
      }
    }
  }
}

// ---------------------------------------------------------------------------
// Flash attention: grid (S/64, B*H), block 128 (4 waves), 16-query tile/wave.
//   scores = (Q*K^T)/128 + mask[k]*-1e9 ; online softmax ; ctx = P*V.
// Q,K bf16 [B,H,S,D]; V bf16 [B,H,D,S]; ctx bf16 [B,S,E].
// Double-buffered async K/V chunk staging.
// ---------------------------------------------------------------------------
__global__ __launch_bounds__(128) void attn_wmma(
    const __bf16* __restrict__ Q, const __bf16* __restrict__ Kb,
    const __bf16* __restrict__ Vb, const float* __restrict__ mask,
    __bf16* __restrict__ ctx) {
  __shared__ __align__(16) __bf16 Kc[2][32][128];   // [key][d]
  __shared__ __align__(16) __bf16 Vt[2][128][32];   // [d][key]
  __shared__ __align__(16) __bf16 Ps[4][16][32];    // per-wave P patch
  __shared__ float mask_s[2][32];

  int tid = threadIdx.x, lane = tid & 31, wid = tid >> 5;
  int bh = blockIdx.y;  // b*16 + h
  int b = bh >> 4, h = bh & 15;
  int q0 = blockIdx.x * 64 + wid * 16;
  int lh = lane >> 4, nl = lane & 15;

  const size_t headBase = (size_t)bh * 2048 * 128;
  unsigned ldsK[2] = { lds_addr(&Kc[0][0][0]), lds_addr(&Kc[1][0][0]) };
  unsigned ldsV[2] = { lds_addr(&Vt[0][0][0]), lds_addr(&Vt[1][0][0]) };

  auto stage = [&](int buf, int kc) {
    const __bf16* kg = Kb + headBase + (size_t)kc * 128;
    const __bf16* vg = Vb + headBase + kc;  // [B,H,D,S]: + d*2048 + key
#pragma unroll
    for (int t = 0; t < 4; ++t) {
      int idx = tid + t * 128;
      async_copy16(ldsK[buf] + idx * 16, kg + (size_t)idx * 8);
      int row = idx >> 2, seg = idx & 3;  // row = d, seg*8 = key offset
      async_copy16(ldsV[buf] + idx * 16, vg + (size_t)row * 2048 + seg * 8);
    }
    if (tid < 32) mask_s[buf][tid] = mask[b * 2048 + kc + tid];
  };
  stage(0, 0);

  // Q fragments for this wave's 16 rows (contraction depth 128 in 4 chunks).
  v16bf qf[4];
#pragma unroll
  for (int kk = 0; kk < 4; ++kk)
    qf[kk] = load_frag_row16(Q + headBase + (size_t)q0 * 128 + kk * 32, 128, lane);

  v8f acc[8] = {};
  float mrow[8], lrow[8];
#pragma unroll
  for (int r = 0; r < 8; ++r) { mrow[r] = -3.0e38f; lrow[r] = 0.0f; }
  const float inv = 1.0f / 128.0f;

  for (int s = 0; s < 64; ++s) {
    int cur = s & 1;
    wait_async0();
    __syncthreads();
    if (s + 1 < 64) stage(cur ^ 1, (s + 1) * 32);

    // Two 16x16 score tiles (keys kc..kc+15, kc+16..kc+31).
    v8f s0 = {}, s1 = {};
#pragma unroll
    for (int kk = 0; kk < 4; ++kk) {
      v16bf k0f = load_frag_row16(&Kc[cur][0][kk * 32], 128, lane);   // K^T
      v16bf k1f = load_frag_row16(&Kc[cur][16][kk * 32], 128, lane);
      s0 = WMMA_BF16(qf[kk], k0f, s0);
      s1 = WMMA_BF16(qf[kk], k1f, s1);
    }
    float madd0 = mask_s[cur][nl] * -1e9f;
    float madd1 = mask_s[cur][16 + nl] * -1e9f;
    float cm[8];
#pragma unroll
    for (int r = 0; r < 8; ++r) {
      s0[r] = s0[r] * inv + madd0;
      s1[r] = s1[r] * inv + madd1;
      cm[r] = fmaxf(s0[r], s1[r]);
    }
#pragma unroll
    for (int off = 1; off < 16; off <<= 1)
#pragma unroll
      for (int r = 0; r < 8; ++r)
        cm[r] = fmaxf(cm[r], __shfl_xor(cm[r], off, 32));

    float cs[8];
#pragma unroll
    for (int r = 0; r < 8; ++r) {
      float mnew  = fmaxf(mrow[r], cm[r]);
      float alpha = __expf(mrow[r] - mnew);
      mrow[r] = mnew;
      float p0 = __expf(s0[r] - mnew);
      float p1 = __expf(s1[r] - mnew);
      int m = r + 8 * lh;
      Ps[wid][m][nl]      = f2bf(p0);
      Ps[wid][m][16 + nl] = f2bf(p1);
      cs[r] = p0 + p1;
      lrow[r] *= alpha;
#pragma unroll
      for (int dt = 0; dt < 8; ++dt) acc[dt][r] *= alpha;
    }
#pragma unroll
    for (int off = 1; off < 16; off <<= 1)
#pragma unroll
      for (int r = 0; r < 8; ++r)
        cs[r] += __shfl_xor(cs[r], off, 32);
#pragma unroll
    for (int r = 0; r < 8; ++r) lrow[r] += cs[r];

    // P (16x32) * V (32x128), V fragments contiguous from Vt[d][key].
    v16bf pf = load_frag_row16(&Ps[wid][0][0], 32, lane);
#pragma unroll
    for (int dt = 0; dt < 8; ++dt) {
      v16bf vf = load_frag_row16(&Vt[cur][dt * 16][0], 32, lane);
      acc[dt] = WMMA_BF16(pf, vf, acc[dt]);
    }
  }

  // Normalize and write ctx[b, q, h*128 + d] in bf16.
#pragma unroll
  for (int dt = 0; dt < 8; ++dt)
#pragma unroll
    for (int r = 0; r < 8; ++r) {
      int m = r + 8 * lh;
      size_t q = (size_t)q0 + m;
      float v = acc[dt][r] / lrow[r];
      ctx[((size_t)b * 2048 + q) * 2048 + h * 128 + dt * 16 + nl] = f2bf(v);
    }
}

// ---------------------------------------------------------------------------
// Launch.  Workspace layout (bytes):
//   [0)          x_bf16      16,777,216   [B*S,2048]
//   [16777216)   WqkvT_bf16  25,165,824   [6144,2048]
//   [41943040)   WoutT_bf16   8,388,608   [2048,2048]
//   [50331648)   Q           16,777,216   [B,H,S,D]
//   [67108864)   K           16,777,216   [B,H,S,D]
//   [83886080)   V           16,777,216   [B,H,D,S]
//   [100663296)  ctx         16,777,216   [B,S,E]     (total ~112 MB)
// ---------------------------------------------------------------------------
extern "C" void kernel_launch(void* const* d_in, const int* in_sizes, int n_in,
                              void* d_out, int out_size, void* d_ws, size_t ws_size,
                              hipStream_t stream) {
  const float* x    = (const float*)d_in[0];
  const float* mask = (const float*)d_in[1];
  const float* Wqkv = (const float*)d_in[2];
  const float* bqkv = (const float*)d_in[3];
  const float* Wout = (const float*)d_in[4];
  const float* bout = (const float*)d_in[5];
  float* out = (float*)d_out;

  char* ws = (char*)d_ws;
  __bf16* xb    = (__bf16*)(ws);
  __bf16* wqkvT = (__bf16*)(ws + 16777216);
  __bf16* woutT = (__bf16*)(ws + 41943040);
  __bf16* qb    = (__bf16*)(ws + 50331648);
  __bf16* kb    = (__bf16*)(ws + 67108864);
  __bf16* vb    = (__bf16*)(ws + 83886080);
  __bf16* ctx   = (__bf16*)(ws + 100663296);

  cvt4<<<dim3(2097152 / 256), 256, 0, stream>>>((const f32x4*)x, (u32x2*)xb, 2097152);
  cvtT<<<dim3(6144 / 32, 2048 / 32), 256, 0, stream>>>(Wqkv, wqkvT, 2048, 6144);
  cvtT<<<dim3(2048 / 32, 2048 / 32), 256, 0, stream>>>(Wout, woutT, 2048, 2048);

  // QKV projection -> Q/K ([B,H,S,D]) and V ([B,H,D,S])
  gemm_bf16_wmma<<<dim3(6144 / 128, 4096 / 128), 256, 0, stream>>>(
      xb, wqkvT, bqkv, nullptr, qb, kb, vb, 4096, 6144, 2048, 1);

  // Flash attention -> ctx bf16 [B,S,E]
  attn_wmma<<<dim3(2048 / 64, 32), 128, 0, stream>>>(qb, kb, vb, mask, ctx);

  // Output projection -> fp32 d_out
  gemm_bf16_wmma<<<dim3(2048 / 128, 4096 / 128), 256, 0, stream>>>(
      ctx, woutT, bout, out, nullptr, nullptr, nullptr, 4096, 2048, 2048, 0);
}
